// P4M_GroupConv2d_Ref_78993038508601
// MI455X (gfx1250) — matrix operand order repaired and checked by reading, not compiled, not run on hardware
//
#include <hip/hip_runtime.h>
#include <hip/hip_bf16.h>

// ---------------------------------------------------------------------------
// P4M group conv as implicit GEMM on CDNA5 WMMA (bf16 inputs, f32 accum).
//   M = 8*Cout = 512, N = B*H*W = 65536, K = Cin*8*9 = 4608
// Kernel 1: build group-transformed weight bank in WMMA-A swizzled layout (ws)
// Kernel 2: tiled conv, LDS-staged x (bf16), v_wmma_f32_16x16x32_bf16
//   Each wave: 32 M rows x 64 N cols -> B fragments reused across 2 WMMAs,
//   and only 2 M-groups -> x HBM redundancy 6x instead of 12x.
// ---------------------------------------------------------------------------

typedef __bf16 bf16_t;
typedef __attribute__((ext_vector_type(16))) __bf16 v16bf;
typedef __attribute__((ext_vector_type(8)))  __bf16 v8bf;
typedef __attribute__((ext_vector_type(8)))  float  v8f;

#define BATCH 16
#define CIN   64
#define COUT  64
#define NG    8
#define HH    64
#define WW    64
#define CI8   (CIN*NG)    // 512 (K channels per tap)
#define OC8   (NG*COUT)   // 512 (M)
#define NTAP  9

// U[g][gin] = COMP[INV[g]][gin]  (p4m composition, precomputed)
__device__ __constant__ int U_TAB[8][8] = {
    {0,1,2,3,4,5,6,7},
    {3,0,1,2,7,4,5,6},
    {2,3,0,1,6,7,4,5},
    {1,2,3,0,5,6,7,4},
    {4,7,6,5,0,3,2,1},
    {5,4,7,6,1,0,3,2},
    {6,5,4,7,2,1,0,3},
    {7,6,5,4,3,2,1,0}};

// ---------------------------------------------------------------------------
// Bank layout (bf16 in d_ws):
//   flat = ((t*32 + mb)*16 + kb)*512 + lane*16 + e
//   mb = oc8/16 (0..31), kb = ci8/32 (0..15), lane 0..31, e 0..15
// A-fragment mapping (ISA 16-bit A 16x32):
//   m_local = lane&15 ; k_local = e + (e<8?0:8) + (lane<16?0:8)
// ---------------------------------------------------------------------------
__global__ void p4m_prep_bank(const float* __restrict__ wrel,
                              bf16_t* __restrict__ bank) {
    int idx = blockIdx.x * 256 + threadIdx.x;     // total 9*32*16*512 = 2359296
    int e    =  idx        & 15;
    int l    = (idx >> 4)  & 31;
    int kb   = (idx >> 9)  & 15;
    int mb   = (idx >> 13) & 31;
    int t    =  idx >> 18;
    if (t >= NTAP) return;

    int m_loc = l & 15;
    int k_loc = e + (e < 8 ? 0 : 8) + (l < 16 ? 0 : 8);
    int oc8 = mb * 16 + m_loc;     // g*COUT + cout
    int ci8 = kb * 32 + k_loc;     // cin*8 + gin
    int g    = oc8 >> 6, cout = oc8 & 63;
    int cin  = ci8 >> 3, gin  = ci8 & 7;
    int kh = t / 3, kw = t % 3;

    // inverse spatial transform: Wt = rot90^r(flipW^m(W))
    int r = g & 3, mfl = g >> 2;
    int si = kh, sj = kw;
    #pragma unroll
    for (int it = 0; it < 3; ++it) {
        if (it < r) { int ti = si; si = sj; sj = 2 - ti; }
    }
    if (mfl) sj = 2 - sj;
    int u = U_TAB[g][gin];

    float v = wrel[(((cout * CIN + cin) * NG + u) * 3 + si) * 3 + sj];
    bank[idx] = (bf16_t)v;
}

// ---------------------------------------------------------------------------
// Conv kernel. grid = (h=64, b=16, mg=2). 256 threads = 8 waves.
// Wave w owns M rows [mg*256 + w*32 , +32) (two A tiles), N = full 64-wide row.
// LDS: x chunk (32 ch) x 3 rows x 66 cols, layout [row][col][k pad 40] bf16.
// ---------------------------------------------------------------------------
#define KPAD 40
#define XS_ELEMS (3 * 66 * KPAD)

__global__ void __launch_bounds__(256, 2)
p4m_conv_wmma(const float* __restrict__ x,
              const float* __restrict__ bias,
              const bf16_t* __restrict__ bank,
              float* __restrict__ out) {
    const int h    = blockIdx.x;
    const int b    = blockIdx.y;
    const int mg   = blockIdx.z;
    const int tid  = threadIdx.x;
    const int wave = tid >> 5;
    const int lane = tid & 31;

    __shared__ bf16_t xs[XS_ELEMS];

    v8f a0s0 = {}, a0s1 = {}, a0s2 = {}, a0s3 = {};   // accum for M tile 0
    v8f a1s0 = {}, a1s1 = {}, a1s2 = {}, a1s3 = {};   // accum for M tile 1

    const int n_loc = lane & 15;
    const int khalf = (lane < 16) ? 0 : 8;    // B-fragment k offset
    const int mb0   = mg * 16 + wave * 2;     // A M-block indices (0..31)
    const int mb1   = mb0 + 1;

    #pragma unroll 1
    for (int chunk = 0; chunk < 16; ++chunk) {
        __syncthreads();
        // ---- stage x[chunk*32 .. +32) channels, rows h-1..h+1, cols -1..64
        for (int i = tid; i < 3 * 66 * 32; i += 256) {
            int wi = i % 66;
            int r2 = i / 66;
            int kh = r2 % 3;
            int k  = r2 / 3;
            int row = h + kh - 1;
            int w   = wi - 1;
            float v = 0.0f;
            if ((unsigned)row < HH && (unsigned)w < WW)
                v = x[((size_t)(b * CI8 + chunk * 32 + k)) * (HH * WW)
                      + row * WW + w];
            xs[(kh * 66 + wi) * KPAD + k] = (bf16_t)v;
        }
        __syncthreads();

        #pragma unroll
        for (int t = 0; t < NTAP; ++t) {
            const int kh = t / 3, kw = t % 3;
            // A fragments: one contiguous 32B per lane, swizzled in prep.
            const v16bf a0 = *(const v16bf*)(bank +
                (((size_t)(t * 32 + mb0) * 16 + chunk) * 512 + lane * 16));
            const v16bf a1 = *(const v16bf*)(bank +
                (((size_t)(t * 32 + mb1) * 16 + chunk) * 512 + lane * 16));
            #pragma unroll
            for (int s = 0; s < 4; ++s) {
                const int wi  = s * 16 + n_loc + kw;        // input col + 1
                const int off = (kh * 66 + wi) * KPAD + khalf;
                // 16B-aligned halves (KPAD*2B = 80B rows, +16B for lane hi, +32B for hi half)
                v8bf blo = *(const v8bf*)(&xs[off]);
                v8bf bhi = *(const v8bf*)(&xs[off + 16]);
                v16bf bm = __builtin_shufflevector(blo, bhi,
                    0,1,2,3,4,5,6,7,8,9,10,11,12,13,14,15);
                v8f c0 = (s == 0) ? a0s0 : (s == 1) ? a0s1 : (s == 2) ? a0s2 : a0s3;
                v8f c1 = (s == 0) ? a1s0 : (s == 1) ? a1s1 : (s == 2) ? a1s2 : a1s3;
                c0 = __builtin_amdgcn_wmma_f32_16x16x32_bf16(
                        false, a0, false, bm, (short)0, c0, false, false);
                c1 = __builtin_amdgcn_wmma_f32_16x16x32_bf16(
                        false, a1, false, bm, (short)0, c1, false, false);
                if (s == 0) { a0s0 = c0; a1s0 = c1; }
                else if (s == 1) { a0s1 = c0; a1s1 = c1; }
                else if (s == 2) { a0s2 = c0; a1s2 = c1; }
                else { a0s3 = c0; a1s3 = c1; }
            }
        }
    }

    // ---- epilogue: C layout VGPR j -> M = j + (lane>=16 ? 8 : 0), N = lane&15
    const int m_hi = (lane < 16) ? 0 : 8;
    #pragma unroll
    for (int mt = 0; mt < 2; ++mt) {
        #pragma unroll
        for (int j = 0; j < 8; ++j) {
            int oc8  = (mb0 + mt) * 16 + j + m_hi;
            int g    = oc8 >> 6;
            int cout = oc8 & 63;
            float bv = bias[cout];
            size_t base = ((size_t)((b * COUT + cout) * NG + g)) * (HH * WW)
                          + (size_t)h * WW;
            if (mt == 0) {
                out[base +  0 + n_loc] = a0s0[j] + bv;
                out[base + 16 + n_loc] = a0s1[j] + bv;
                out[base + 32 + n_loc] = a0s2[j] + bv;
                out[base + 48 + n_loc] = a0s3[j] + bv;
            } else {
                out[base +  0 + n_loc] = a1s0[j] + bv;
                out[base + 16 + n_loc] = a1s1[j] + bv;
                out[base + 32 + n_loc] = a1s2[j] + bv;
                out[base + 48 + n_loc] = a1s3[j] + bv;
            }
        }
    }
}

extern "C" void kernel_launch(void* const* d_in, const int* in_sizes, int n_in,
                              void* d_out, int out_size, void* d_ws, size_t ws_size,
                              hipStream_t stream) {
    const float* x    = (const float*)d_in[0];   // (16,64,8,64,64)
    const float* wrel = (const float*)d_in[1];   // (64,64,8,3,3)
    const float* bias = (const float*)d_in[2];   // (64,)
    float* out = (float*)d_out;                  // (16,64,8,64,64)
    bf16_t* bank = (bf16_t*)d_ws;                // 9*512*512 bf16 = 4.5 MB

    // Kernel 1: group-transformed, WMMA-swizzled weight bank
    const int prep_elems = NTAP * 32 * 16 * 512;          // 2,359,296
    p4m_prep_bank<<<prep_elems / 256, 256, 0, stream>>>(wrel, bank);

    // Kernel 2: implicit-GEMM conv via bf16 WMMA (2 M-groups, 32 M per wave)
    dim3 grid(HH, BATCH, 2);
    p4m_conv_wmma<<<grid, 256, 0, stream>>>(x, bias, bank, out);
}
